// GraphAttentionLayer_34016140984478
// MI455X (gfx1250) — matrix operand order repaired and checked by reading, not compiled, not run on hardware
//
#include <hip/hip_runtime.h>
#include <hip/hip_bf16.h>

#define ALPHA 0.2f
#define BB 16
#define NN 1024
#define FF 512

typedef __attribute__((ext_vector_type(16))) __bf16 v16bf;
typedef __attribute__((ext_vector_type(8)))  __bf16 v8bf;
typedef __attribute__((ext_vector_type(8)))  float  v8f;

// round-to-nearest-even f32 -> bf16 bit pattern
__device__ __forceinline__ unsigned bfbits(float x) {
    union { float f; unsigned u; } c; c.f = x;
    return (c.u + 0x7FFFu + ((c.u >> 16) & 1u)) >> 16;
}

// ---------------- Kernel 1: w1 = W^T a1, w2 = W^T a2 ----------------
__global__ void gat_prep_w(const float* __restrict__ W, const float* __restrict__ a1,
                           const float* __restrict__ a2, float* __restrict__ w1,
                           float* __restrict__ w2) {
    int f = blockIdx.x * blockDim.x + threadIdx.x;
    if (f >= FF) return;
    float s1 = 0.f, s2 = 0.f;
    for (int o = 0; o < FF; ++o) {
        float wv = W[o * FF + f];        // coalesced across threads
        s1 += wv * a1[o];                // a1/a2 scalar-broadcast
        s2 += wv * a2[o];
    }
    w1[f] = s1; w2[f] = s2;
}

// ---------------- Kernel 2: f1[b,n] = X[b,n,:].w1 ; f2 likewise ------
__global__ void gat_node_scores(const float* __restrict__ X, const float* __restrict__ w1,
                                const float* __restrict__ w2, float* __restrict__ f1,
                                float* __restrict__ f2) {
    int gid  = blockIdx.x * blockDim.x + threadIdx.x;
    int node = gid >> 5;                 // one wave32 per node
    int lane = gid & 31;
    const float4* x4  = (const float4*)(X + (size_t)node * FF);
    const float4* w14 = (const float4*)w1;
    const float4* w24 = (const float4*)w2;
    float s1 = 0.f, s2 = 0.f;
#pragma unroll
    for (int i = 0; i < 4; ++i) {        // 128 float4 per row / 32 lanes
        int q = lane + i * 32;
        float4 xv = x4[q], wa = w14[q], wb = w24[q];
        s1 += xv.x * wa.x + xv.y * wa.y + xv.z * wa.z + xv.w * wa.w;
        s2 += xv.x * wb.x + xv.y * wb.y + xv.z * wb.z + xv.w * wb.w;
    }
#pragma unroll
    for (int off = 16; off > 0; off >>= 1) {
        s1 += __shfl_xor(s1, off, 32);
        s2 += __shfl_xor(s2, off, 32);
    }
    if (lane == 0) { f1[node] = s1; f2[node] = s2; }
}

// ------- Kernel 2.5: one-shot X[B,N,F] f32 -> Xt[B,F,N] bf16 (LDS tiled) -------
__launch_bounds__(256)
__global__ void gat_transpose_bf16(const float* __restrict__ X, unsigned short* __restrict__ Xt) {
    const int b  = blockIdx.z;
    const int n0 = blockIdx.x * 32;
    const int f0 = blockIdx.y * 32;
    const int tid = threadIdx.x;
    __shared__ float tile[32][33];
    // read 32n x 32f, coalesced along f
#pragma unroll
    for (int i = 0; i < 4; ++i) {
        int idx = tid + i * 256;
        int r = idx >> 5, c = idx & 31;
        tile[r][c] = X[((size_t)b * NN + n0 + r) * FF + f0 + c];
    }
    __syncthreads();
    // write 32f x 32n bf16, two n packed per dword, coalesced along n
#pragma unroll
    for (int i = 0; i < 2; ++i) {
        int idx = tid + i * 256;
        int rf  = idx >> 4;              // 0..31 (f)
        int cn  = (idx & 15) * 2;        // 0..30 (n, even)
        unsigned pk = bfbits(tile[cn][rf]) | (bfbits(tile[cn + 1][rf]) << 16);
        *(unsigned*)(&Xt[((size_t)b * FF + f0 + rf) * NN + n0 + cn]) = pk;
    }
}

// ---------------- Kernel 3: fused masked-softmax + WMMA aggregation --
#define KSTRIDE 40   // Pt k-stride (elements): 16B-aligned fragments

__launch_bounds__(256)
__global__ void gat_attn_spmm(const unsigned short* __restrict__ Xt, const int* __restrict__ adj,
                              const float* __restrict__ f1, const float* __restrict__ f2,
                              float* __restrict__ out) {
    const int b    = blockIdx.y;
    const int n0   = blockIdx.x * 16;
    const int tid  = threadIdx.x;
    const int wave = tid >> 5;
    const int lane = tid & 31;
    const int lr   = lane & 15;          // fragment row (A) / col (B)
    const int lg   = lane >> 4;          // lane group

    __shared__ __align__(16) __bf16 Pt[2][16 * KSTRIDE];  // double-buffered weight tile
    __shared__ float rowsum[16];

    if (tid < 16) rowsum[tid] = 0.f;

    // stage-phase role for the P tile: thread -> (row pn, cols pm, pm+1)
    const int pn = tid >> 4;             // 0..15
    const int pm = (tid & 15) * 2;       // 0..30
    const float  f1v    = f1[b * NN + n0 + pn];
    const float  cn     = f1v > 0.f ? f1v : 0.f;     // per-row softmax stabilizer
    const float* f2row  = f2 + b * NN;
    const int*   adjrow = adj + (size_t)b * NN * NN + (size_t)(n0 + pn) * NN;

    // per-wave B-fragment row pointers (bf16 [f][m], contiguous along m)
    const unsigned short* Xb = Xt + (size_t)b * FF * NN;
    const unsigned short* rowp[4];
#pragma unroll
    for (int j = 0; j < 4; ++j)
        rowp[j] = Xb + (size_t)((wave * 4 + j) * 16 + lr) * NN + lg * 16;

    v8f acc[4] = {};

    for (int it = 0; it < NN / 32; ++it) {
        const int m0 = it * 32;
        __bf16* ptb = Pt[it & 1];

        // ---- attention-weight tile P[16 x 32], branchless masked exp ----
        {
            float2 fv = *(const float2*)(&f2row[m0 + pm]);
            int2   av = *(const int2*)(&adjrow[m0 + pm]);
            float e0 = f1v + fv.x;
            float e1 = f1v + fv.y;
            e0 = e0 > 0.f ? e0 : ALPHA * e0;          // leaky_relu
            e1 = e1 > 0.f ? e1 : ALPHA * e1;
            float s0 = av.x ? (e0 - cn) : -100.f;     // exp(-100) ~ 0 == masked
            float s1 = av.y ? (e1 - cn) : -100.f;
            float w0  = __expf(s0);                   // unconditional: no exec branching
            float w1v = __expf(s1);
            *(unsigned*)(&ptb[pn * KSTRIDE + pm]) = bfbits(w0) | (bfbits(w1v) << 16);
            atomicAdd(&rowsum[pn], w0 + w1v);         // ds_add_f32
        }
        __syncthreads();   // single barrier: double-buffered Pt

        // ---- A fragment: 16x32 bf16, contiguous k = lg*16 + s (from LDS) ----
        v16bf afrag;
        {
            v8bf alo = *(const v8bf*)(&ptb[lr * KSTRIDE + lg * 16]);      // K = lg*16 .. +7
            v8bf ahi = *(const v8bf*)(&ptb[lr * KSTRIDE + lg * 16 + 8]);  // K = lg*16+8 .. +15
#pragma unroll
            for (int i = 0; i < 8; ++i) { afrag[i] = alo[i]; afrag[i + 8] = ahi[i]; }
        }

        // ---- B fragments: direct global_load_b128 pairs from pre-transposed bf16 X ----
        v8bf blo[4], bhi[4];
#pragma unroll
        for (int j = 0; j < 4; ++j) {
            blo[j] = *(const v8bf*)(rowp[j] + m0);       // K = lg*16 .. +7
            bhi[j] = *(const v8bf*)(rowp[j] + m0 + 8);   // K = lg*16+8 .. +15
        }
#pragma unroll
        for (int j = 0; j < 4; ++j) {
            v16bf bfrag;
#pragma unroll
            for (int i = 0; i < 8; ++i) { bfrag[i] = blo[j][i]; bfrag[i + 8] = bhi[j][i]; }
            acc[j] = __builtin_amdgcn_wmma_f32_16x16x32_bf16(
                false, afrag, false, bfrag, (short)0, acc[j], false, false);
        }
    }

    // ---- 16 fast reciprocals instead of 8192 divides ----
    __syncthreads();
    if (tid < 16) rowsum[tid] = __builtin_amdgcn_rcpf(fmaxf(rowsum[tid], 1e-30f));
    __syncthreads();

    // ---- epilogue: normalize softmax, ELU, store ----
#pragma unroll
    for (int j = 0; j < 4; ++j) {
        int col = (wave * 4 + j) * 16 + lr;
#pragma unroll
        for (int rr = 0; rr < 8; ++rr) {
            int M = rr + lg * 8;                     // C/D layout: vgpr rr, lane group lg
            float v = acc[j][rr] * rowsum[M];
            v = v > 0.f ? v : __expf(v) - 1.0f;      // ELU (alpha = 1)
            out[((size_t)b * NN + n0 + M) * FF + col] = v;
        }
    }
}

extern "C" void kernel_launch(void* const* d_in, const int* in_sizes, int n_in,
                              void* d_out, int out_size, void* d_ws, size_t ws_size,
                              hipStream_t stream) {
    (void)in_sizes; (void)n_in; (void)out_size; (void)ws_size;
    const float* X   = (const float*)d_in[0];
    const int*   adj = (const int*)d_in[1];
    const float* W   = (const float*)d_in[2];
    const float* a1  = (const float*)d_in[3];
    const float* a2  = (const float*)d_in[4];
    float* out = (float*)d_out;

    unsigned short* Xt = (unsigned short*)d_ws;      // [16*512*1024] bf16 (16.8 MB)
    float* w1 = (float*)(Xt + (size_t)BB * FF * NN); // [512]
    float* w2 = w1 + FF;                             // [512]
    float* f1 = w2 + FF;                             // [16*1024]
    float* f2 = f1 + BB * NN;                        // [16*1024]

    gat_prep_w<<<dim3((FF + 255) / 256), 256, 0, stream>>>(W, a1, a2, w1, w2);
    gat_node_scores<<<dim3(BB * NN / 8), 256, 0, stream>>>(X, w1, w2, f1, f2);
    gat_transpose_bf16<<<dim3(NN / 32, FF / 32, BB), 256, 0, stream>>>(X, Xt);
    gat_attn_spmm<<<dim3(NN / 16, BB), 256, 0, stream>>>(Xt, adj, f1, f2, out);
}
